// SparseMoELayer_87342454931823
// MI455X (gfx1250) — matrix operand order repaired and checked by reading, not compiled, not run on hardware
//
#include <hip/hip_runtime.h>
#include <math.h>

// ---------------- problem constants ----------------
#define N_TOK 32768
#define DIM   512
#define NEXP  8
#define HID   128
#define CAP   5120              // int(1.25 * 32768 / 8)
#define TILE_M 32
#define TILES_PER_E (CAP / TILE_M)   // 160
#define RBLK 128                // router/compact tokens per block
#define NRB (N_TOK / RBLK)      // 256 blocks

typedef _Float16 v16h __attribute__((ext_vector_type(16)));
typedef _Float16 v8h  __attribute__((ext_vector_type(8)));
typedef _Float16 v4h  __attribute__((ext_vector_type(4)));
typedef float    v8f  __attribute__((ext_vector_type(8)));

union ABFrag { v16h v; v8h h[2]; };

__device__ __forceinline__ v8f wmma_f16(v16h a, v16h b, v8f c) {
  // D = A*B + C, f32 accumulate, 16x16x32 f16
  return __builtin_amdgcn_wmma_f32_16x16x32_f16(false, a, false, b, (short)0, c,
                                                false, false);
}

// ============================================================
// K0: convert W1/W2 (f32) -> f16 in WMMA B-fragment-swizzled order.
// Contiguous-k B fragment layout used here (16-bit, 32x16; A uses same order): lane = 16*(K/16) + N%16,
// half j within lane = K%16.  Frag = 512 halfs, fully coalesced per-wave load.
// ============================================================
__global__ __launch_bounds__(256) void moe_convert_kernel(
    const float* __restrict__ W1, const float* __restrict__ W2,
    _Float16* __restrict__ w1f, _Float16* __restrict__ w2f)
{
  int idx = blockIdx.x * 256 + threadIdx.x;   // 2 * 524288 threads
  int which = idx >> 19;
  int r = idx & ((1 << 19) - 1);
  int e  = r >> 16;
  int rr = r & 65535;
  if (which == 0) {               // W1: [E, D, H]  -> B frags (K=D, N=H)
    int d = rr >> 7, h = rr & 127;
    int kc = d >> 5, kk = d & 31;
    int nt = h >> 4, col = h & 15;
    int lane = ((kk >> 4) << 4) | col;
    int j = kk & 15;
    w1f[(size_t)((e * 16 + kc) * 8 + nt) * 512 + lane * 16 + j] = (_Float16)W1[r];
  } else {                        // W2: [E, H, D]  -> B frags (K=H, N=D)
    int h = rr >> 9, d = rr & 511;
    int kc = h >> 5, kk = h & 31;
    int nt = d >> 4, col = d & 15;
    int lane = ((kk >> 4) << 4) | col;
    int j = kk & 15;
    w2f[(size_t)((e * 4 + kc) * 32 + nt) * 512 + lane * 16 + j] = (_Float16)W2[r];
  }
}

// ============================================================
// K1: router in exact fp32; top-2 with lax.top_k tie-break (lowest index).
// Emits packed (e1|e2), per-block expert counts, per-block top1 sums.
// ============================================================
__global__ __launch_bounds__(RBLK) void moe_router_kernel(
    const float* __restrict__ x, const float* __restrict__ Wr,
    const float* __restrict__ br, int* __restrict__ topPair,
    int* __restrict__ blockCounts, float* __restrict__ blockConf)
{
  __shared__ float sWr[DIM * NEXP];   // 16 KB
  __shared__ int   scnt[NEXP];
  __shared__ float sred[RBLK];

  const int t = threadIdx.x, b = blockIdx.x;
  for (int i = t; i < DIM * NEXP; i += RBLK) sWr[i] = Wr[i];
  if (t < NEXP) scnt[t] = 0;
  __syncthreads();

  const int n = b * RBLK + t;
  float acc[NEXP];
#pragma unroll
  for (int e = 0; e < NEXP; ++e) acc[e] = br[e];

  const float4* xr = (const float4*)(x + (size_t)n * DIM);
  for (int i = 0; i < DIM / 4; ++i) {
    float4 v = xr[i];
    const float* w = &sWr[i * 4 * NEXP];
#pragma unroll
    for (int e = 0; e < NEXP; ++e)
      acc[e] += v.x * w[e] + v.y * w[NEXP + e] + v.z * w[2 * NEXP + e] +
                v.w * w[3 * NEXP + e];
  }

  // top-1 (first occurrence of max == lowest index on tie)
  int e1 = 0; float v1 = acc[0];
#pragma unroll
  for (int e = 1; e < NEXP; ++e) if (acc[e] > v1) { v1 = acc[e]; e1 = e; }
  // top-2 excluding e1
  int e2 = (e1 == 0) ? 1 : 0; float v2 = acc[e2];
#pragma unroll
  for (int e = 0; e < NEXP; ++e)
    if (e != e1 && acc[e] > v2) { v2 = acc[e]; e2 = e; }

  topPair[n] = e1 | (e2 << 4);
  atomicAdd(&scnt[e1], 1);
  atomicAdd(&scnt[e2], 1);
  sred[t] = v1;
  __syncthreads();
  for (int s = RBLK / 2; s > 0; s >>= 1) {      // deterministic tree reduce
    if (t < s) sred[t] += sred[t + s];
    __syncthreads();
  }
  if (t == 0) blockConf[b] = sred[0];
  if (t < NEXP) blockCounts[b * NEXP + t] = scnt[t];
}

// ============================================================
// K2: exclusive scan of block counts (token order), load = min(count, CAP),
// writes load_loss / load_dist / confidence into the tail of d_out.
// ============================================================
__global__ void moe_stats_kernel(const int* __restrict__ blockCounts,
                                 const float* __restrict__ blockConf,
                                 int* __restrict__ blockOffsets,
                                 int* __restrict__ loadArr,
                                 float* __restrict__ outTail)
{
  __shared__ int sload[NEXP];
  const int t = threadIdx.x;
  if (t < NEXP) {
    int run = 0;
    for (int b = 0; b < NRB; ++b) {
      blockOffsets[b * NEXP + t] = run;
      run += blockCounts[b * NEXP + t];
    }
    int ld = run < CAP ? run : CAP;
    sload[t] = ld;
    loadArr[t] = ld;
  }
  __syncthreads();
  if (t == 0) {
    float conf = 0.f;
    for (int b = 0; b < NRB; ++b) conf += blockConf[b];
    float sum = 0.f;
    for (int e = 0; e < NEXP; ++e) sum += (float)sload[e];
    float denom = sum + 1e-8f;
    float loss = 0.f;
    for (int e = 0; e < NEXP; ++e) {
      float dist = (float)sload[e] / denom;
      outTail[1 + e] = dist;
      loss += dist * logf(dist + 1e-8f);
    }
    outTail[0] = loss;
    outTail[1 + NEXP] = conf / (float)N_TOK;
  }
}

// ============================================================
// K3: capacity-ordered compaction.  Rank of token n within expert e's queue
// (token order) = blockOffset + preceding-wave totals + intra-wave popc rank.
// Exactly reproduces the reference cumsum/keep semantics; deterministic.
// ============================================================
__global__ __launch_bounds__(RBLK) void moe_compact_kernel(
    const int* __restrict__ topPair, const int* __restrict__ blockOffsets,
    int* __restrict__ tokenList)
{
  __shared__ int wtot[RBLK / 32][NEXP];
  const int t = threadIdx.x, b = blockIdx.x;
  const int n = b * RBLK + t;
  const int p = topPair[n];
  const int e1 = p & 15, e2 = (p >> 4) & 15;
  const int lane = t & 31, w = t >> 5;
  const unsigned long long lmask = (1ull << lane) - 1ull;

  int intra[NEXP], mm[NEXP];
#pragma unroll
  for (int e = 0; e < NEXP; ++e) {
    int m = (e1 == e) || (e2 == e);
    unsigned long long bal = __ballot(m);
    intra[e] = (int)__popcll(bal & lmask);
    mm[e] = m;
    if (lane == 0) wtot[w][e] = (int)__popcll(bal);
  }
  __syncthreads();
#pragma unroll
  for (int e = 0; e < NEXP; ++e) {
    if (!mm[e]) continue;
    int base = blockOffsets[b * NEXP + e];
#pragma unroll
    for (int ww = 0; ww < RBLK / 32; ++ww)
      if (ww < w) base += wtot[ww][e];
    int rank = base + intra[e];
    if (rank < CAP) tokenList[e * CAP + rank] = n;
  }
}

// ============================================================
// K4: expert FFN.  One block = (expert, 32-token tile), 256 threads = 8 waves.
// GEMM1: [32,512]x[512,128] -> relu -> f16 h in LDS.
// GEMM2: [32,128]x[128,512] -> atomicAdd(out) of y+b2 for kept rows.
// A frags from padded LDS; B frags coalesced from pre-swizzled f16 weights.
// ============================================================
__global__ __launch_bounds__(256) void moe_ffn_kernel(
    const float* __restrict__ x, const float* __restrict__ b1,
    const float* __restrict__ b2, const _Float16* __restrict__ w1f,
    const _Float16* __restrict__ w2f, const int* __restrict__ tokenList,
    const int* __restrict__ loadArr, float* __restrict__ out)
{
  const int e    = blockIdx.x / TILES_PER_E;
  const int tile = blockIdx.x % TILES_PER_E;
  const int loadE = loadArr[e];
  const int t0 = tile * TILE_M;
  if (t0 >= loadE) return;

  __shared__ _Float16 sX[TILE_M][DIM + 8];   // +8 halfs: bank-conflict pad
  __shared__ _Float16 sH[TILE_M][HID + 8];
  __shared__ int sTok[TILE_M];

  const int t = threadIdx.x;

  // ---- stage gathered x rows, f32 -> f16 ----
  {
    const int r  = t >> 3;        // 32 rows, 8 threads each
    const int c8 = t & 7;
    int slot = t0 + r;
    int tok = (slot < loadE) ? tokenList[e * CAP + slot] : -1;
    if (c8 == 0) sTok[r] = tok;
    const float4* xr = (tok >= 0) ? (const float4*)(x + (size_t)tok * DIM)
                                  : (const float4*)0;
#pragma unroll
    for (int k = 0; k < 16; ++k) {
      int col = c8 * 4 + k * 32;
      float4 v = xr ? xr[col >> 2] : make_float4(0.f, 0.f, 0.f, 0.f);
      v4h hv = {(_Float16)v.x, (_Float16)v.y, (_Float16)v.z, (_Float16)v.w};
      *(v4h*)&sX[r][col] = hv;
    }
  }
  __syncthreads();

  const int lane  = t & 31;
  const int wave  = t >> 5;
  const int lhalf = lane >> 4;      // A: kbase = 16*lhalf; C/D: M += 8*lhalf
  const int lcol  = lane & 15;      // A row / C-D column
  const int m0 = (wave & 1) * 16;
  const int np = wave >> 1;         // 0..3

  // ---- GEMM1: h = relu(x @ W1_e + b1_e), N-tiles {2np, 2np+1} ----
  v8f acc0 = {}; v8f acc1 = {};
  for (int kc = 0; kc < DIM / 32; ++kc) {
    ABFrag a;
    const int kb = kc * 32 + lhalf * 16;
    a.h[0] = *(const v8h*)&sX[m0 + lcol][kb];        // K = kb .. kb+7
    a.h[1] = *(const v8h*)&sX[m0 + lcol][kb + 8];    // K = kb+8 .. kb+15
    v16h bf0 = *(const v16h*)(w1f +
        (size_t)((e * 16 + kc) * 8 + np * 2 + 0) * 512 + lane * 16);
    v16h bf1 = *(const v16h*)(w1f +
        (size_t)((e * 16 + kc) * 8 + np * 2 + 1) * 512 + lane * 16);
    acc0 = wmma_f16(a.v, bf0, acc0);
    acc1 = wmma_f16(a.v, bf1, acc1);
  }
  {
    const int n0 = np * 32;
    const float bb0 = b1[e * HID + n0 + lcol];
    const float bb1 = b1[e * HID + n0 + 16 + lcol];
#pragma unroll
    for (int r = 0; r < 8; ++r) {
      int mrow = m0 + r + 8 * lhalf;
      float h0 = acc0[r] + bb0; h0 = h0 > 0.f ? h0 : 0.f;
      float h1 = acc1[r] + bb1; h1 = h1 > 0.f ? h1 : 0.f;
      sH[mrow][n0 + lcol]      = (_Float16)h0;
      sH[mrow][n0 + 16 + lcol] = (_Float16)h1;
    }
  }
  __syncthreads();

  // ---- GEMM2: y = h @ W2_e + b2_e ; scatter-add kept rows ----
  ABFrag a2[HID / 32];
#pragma unroll
  for (int kc = 0; kc < HID / 32; ++kc) {
    const int kb = kc * 32 + lhalf * 16;
    a2[kc].h[0] = *(const v8h*)&sH[m0 + lcol][kb];
    a2[kc].h[1] = *(const v8h*)&sH[m0 + lcol][kb + 8];
  }
  for (int i = 0; i < 8; ++i) {
    const int nt = np * 8 + i;          // 32 N-tiles over D=512
    v8f acc = {};
#pragma unroll
    for (int kc = 0; kc < HID / 32; ++kc) {
      v16h bf = *(const v16h*)(w2f +
          (size_t)((e * 4 + kc) * 32 + nt) * 512 + lane * 16);
      acc = wmma_f16(a2[kc].v, bf, acc);
    }
    const int n = nt * 16 + lcol;
    const float bb = b2[e * DIM + n];
#pragma unroll
    for (int r = 0; r < 8; ++r) {
      int mrow = m0 + r + 8 * lhalf;
      if (t0 + mrow < loadE) {
        int tok = sTok[mrow];
        // <=2 contributions per element onto exact zero: order-independent.
        atomicAdd(out + (size_t)tok * DIM + n, acc[r] + bb);
      }
    }
  }
}

// ============================================================
extern "C" void kernel_launch(void* const* d_in, const int* in_sizes, int n_in,
                              void* d_out, int out_size, void* d_ws, size_t ws_size,
                              hipStream_t stream)
{
  (void)in_sizes; (void)n_in; (void)out_size; (void)ws_size;
  const float* x  = (const float*)d_in[0];
  const float* Wr = (const float*)d_in[1];
  const float* br = (const float*)d_in[2];
  const float* W1 = (const float*)d_in[3];
  const float* b1 = (const float*)d_in[4];
  const float* W2 = (const float*)d_in[5];
  const float* b2 = (const float*)d_in[6];
  float* out = (float*)d_out;

  // ---- workspace carve-up (~2.4 MB total) ----
  char* ws = (char*)d_ws;
  size_t off = 0;
  _Float16* w1f = (_Float16*)(ws + off); off += (size_t)NEXP * DIM * HID * 2; // 1 MB
  _Float16* w2f = (_Float16*)(ws + off); off += (size_t)NEXP * HID * DIM * 2; // 1 MB
  int*   tokenList    = (int*)(ws + off); off += (size_t)NEXP * CAP * 4;      // 160 KB
  int*   topPair      = (int*)(ws + off); off += (size_t)N_TOK * 4;           // 128 KB
  int*   blockCounts  = (int*)(ws + off); off += (size_t)NRB * NEXP * 4;
  int*   blockOffsets = (int*)(ws + off); off += (size_t)NRB * NEXP * 4;
  float* blockConf    = (float*)(ws + off); off += (size_t)NRB * 4;
  int*   loadArr      = (int*)(ws + off); off += 64;

  // out tokens region must start at zero (atomic accumulation target)
  hipMemsetAsync(d_out, 0, (size_t)N_TOK * DIM * sizeof(float), stream);

  moe_convert_kernel<<<dim3(2 * NEXP * DIM * HID / 256), dim3(256), 0, stream>>>(
      W1, W2, w1f, w2f);

  moe_router_kernel<<<dim3(NRB), dim3(RBLK), 0, stream>>>(
      x, Wr, br, topPair, blockCounts, blockConf);

  moe_stats_kernel<<<dim3(1), dim3(32), 0, stream>>>(
      blockCounts, blockConf, blockOffsets, loadArr,
      out + (size_t)N_TOK * DIM);

  moe_compact_kernel<<<dim3(NRB), dim3(RBLK), 0, stream>>>(
      topPair, blockOffsets, tokenList);

  moe_ffn_kernel<<<dim3(NEXP * TILES_PER_E), dim3(256), 0, stream>>>(
      x, b1, b2, w1f, w2f, tokenList, loadArr, out);
}
